// GNNLayer_13657996002096
// MI455X (gfx1250) — matrix operand order repaired and checked by reading, not compiled, not run on hardware
//
#include <hip/hip_runtime.h>
#include <cstdint>

// ---------------------------------------------------------------------------
// MI455X (gfx1250) GNN layer: split-bf16 (hi/lo) WMMA GEMMs, fp32 accumulate.
// D = A*B via 3x v_wmma_f32_16x16x32_bf16 (hi*hi + hi*lo + lo*hi) ~= fp32.
// ---------------------------------------------------------------------------

typedef __attribute__((ext_vector_type(16))) __bf16 v16bf;
typedef __attribute__((ext_vector_type(8)))  __bf16 v8bf;
typedef __attribute__((ext_vector_type(8)))  float  v8f;

#define HDIM 128
#define HSTRIDE 136   // staged-h row stride in bf16 (128 + pad): 272B = 68 dw == 4 mod 64 -> conflict-free
#define USTRIDE 264   // staged-u row stride in bf16 (256 + pad): 528B = 132 dw == 4 mod 64

__device__ __forceinline__ v8f wmma_bf16(v16bf a, v16bf b, v8f c) {
  // (neg_a, A, neg_b, B, c_mod, C, reuse_a, reuse_b)
  return __builtin_amdgcn_wmma_f32_16x16x32_bf16(false, a, false, b, (short)0, c, false, false);
}

__device__ __forceinline__ v8f mm3(v16bf ah, v16bf al, v16bf bh, v16bf bl, v8f c) {
  c = wmma_bf16(ah, bh, c);
  c = wmma_bf16(ah, bl, c);
  c = wmma_bf16(al, bh, c);
  return c;
}

__device__ __forceinline__ void split16(const float* f, v16bf& hi, v16bf& lo) {
#pragma unroll
  for (int q = 0; q < 16; ++q) {
    __bf16 h = (__bf16)f[q];
    hi[q] = h;
    lo[q] = (__bf16)(f[q] - (float)h);
  }
}

__device__ __forceinline__ void load16(const float* src, int cb, float* f) {
  // 16-bit A-matrix layout: per lane two contiguous 8-elem K-chunks at cb and cb+16
  float4 a = *(const float4*)(src + cb);
  float4 b = *(const float4*)(src + cb + 4);
  float4 c = *(const float4*)(src + cb + 16);
  float4 d = *(const float4*)(src + cb + 20);
  f[0]=a.x; f[1]=a.y; f[2]=a.z; f[3]=a.w;
  f[4]=b.x; f[5]=b.y; f[6]=b.z; f[7]=b.w;
  f[8]=c.x; f[9]=c.y; f[10]=c.z; f[11]=c.w;
  f[12]=d.x; f[13]=d.y; f[14]=d.z; f[15]=d.w;
}

__device__ __forceinline__ v16bf cat8(v8bf a, v8bf b) {
  return __builtin_shufflevector(a, b, 0,1,2,3,4,5,6,7,8,9,10,11,12,13,14,15);
}

__device__ __forceinline__ float silu_f(float v) { return v / (1.0f + __expf(-v)); }

// --------------------------- prep kernels ----------------------------------

__global__ void zero_f32(float* __restrict__ p, long long n) {
  long long i = (long long)blockIdx.x * blockDim.x + threadIdx.x;
  if (i < n) p[i] = 0.0f;
}

__global__ void pack_weights(const float* __restrict__ m1, const float* __restrict__ m2,
                             const float* __restrict__ u1, const float* __restrict__ u2,
                             __bf16* m1h, __bf16* m1l, __bf16* m2h, __bf16* m2l,
                             __bf16* u1h, __bf16* u1l, __bf16* u2h, __bf16* u2l,
                             int n1, int n2, int n3, int n4) {
  int i = blockIdx.x * blockDim.x + threadIdx.x;
  const float* s; __bf16 *dh, *dl; int j = i; int nc = HDIM;
  if (j < n1)                    { s = m1; dh = m1h; dl = m1l; }
  else if ((j -= n1) < n2)       { s = m2; dh = m2h; dl = m2l; }
  else if ((j -= n2) < n3)       { s = u1; dh = u1h; dl = u1l; nc = 2 * HDIM; }
  else if ((j -= n3) < n4)       { s = u2; dh = u2h; dl = u2l; }
  else return;
  const int r = j / nc, cc = j % nc;
  const int ln = r & 31, hh = ln >> 4, sq = cc & 15;
  const int kk = (r & ~31) + (sq < 8 ? 8 * hh + sq : 16 + 8 * hh + (sq - 8));
  const int nn = (cc & ~15) + (ln & 15);
  float v = s[kk * nc + nn];
  __bf16 h = (__bf16)v;
  dh[j] = h;
  dl[j] = (__bf16)(v - (float)h);
}

__global__ void degree_kernel(const int* __restrict__ recv, float* __restrict__ cnt, int E) {
  int e = blockIdx.x * blockDim.x + threadIdx.x;
  if (e < E) unsafeAtomicAdd(&cnt[recv[e]], 1.0f);
}

// --------------------------- message kernel --------------------------------
// One wave per (16-edge tile, batch). feats = [x[send] | x[recv] | edge_attr],
// K = 3H = 384 = 12 K-steps of 32. h staged per-wave in LDS as bf16 hi/lo.

__global__ __launch_bounds__(128) void msg_kernel(
    const float* __restrict__ x, const float* __restrict__ ea,
    const int* __restrict__ send, const int* __restrict__ recv,
    const __bf16* __restrict__ m1h, const __bf16* __restrict__ m1l,
    const __bf16* __restrict__ m2h, const __bf16* __restrict__ m2l,
    const float* __restrict__ b1, const float* __restrict__ b2,
    float* __restrict__ msgOut, float* __restrict__ agg,
    int N, int E, int B_, int numTiles) {
  __shared__ __bf16 lds[4][2][16 * HSTRIDE];

  const int wave = threadIdx.x >> 5;
  const int lane = threadIdx.x & 31;
  const int wid  = blockIdx.x * 4 + wave;
  if (wid >= numTiles * B_) return;
  const int tile = wid % numTiles;
  const int b    = wid / numTiles;

  const int m     = lane & 15;        // A-row / C-col within tile
  const int base8 = (lane >> 4) * 8;  // lane-half K/M offset
  const int e     = tile * 16 + m;
  const int ec    = (e < E) ? e : (E - 1);

  const int si = send[ec];
  const int ri = recv[ec];
  const float* xs = x  + ((size_t)b * N + si) * HDIM;
  const float* xr = x  + ((size_t)b * N + ri) * HDIM;
  const float* ae = ea + ((size_t)b * E + ec) * HDIM;

  // Prefetch next tile's edge features (L2 streaming hint).
  {
    int pn = ec + 16 < E ? ec + 16 : ec;
    __builtin_prefetch(ea + ((size_t)b * E + pn) * HDIM, 0, 1);
  }

  v8f zero = {0.f, 0.f, 0.f, 0.f, 0.f, 0.f, 0.f, 0.f};
  v8f c[8];
#pragma unroll
  for (int t = 0; t < 8; ++t) c[t] = zero;

  // ---- layer 1: (16 x 384) @ (384 x 128) ----
#pragma unroll
  for (int ks = 0; ks < 12; ++ks) {
    const float* src = (ks < 4) ? xs : (ks < 8 ? xr : ae);
    const int cb = (ks & 3) * 32 + base8;
    float f[16];
    load16(src, cb, f);
    v16bf ah, al;
    split16(f, ah, al);
    const __bf16* bh = m1h + (size_t)(ks * 32 + lane) * HDIM;
    const __bf16* bl = m1l + (size_t)(ks * 32 + lane) * HDIM;
#pragma unroll
    for (int t = 0; t < 8; ++t) {
      v16bf wbh = *(const v16bf*)(bh + t * 16);
      v16bf wbl = *(const v16bf*)(bl + t * 16);
      c[t] = mm3(ah, al, wbh, wbl, c[t]);
    }
  }

  // ---- bias + SiLU, stage h as bf16 hi/lo in per-wave LDS ----
  __bf16* h_hi = lds[wave][0];
  __bf16* h_lo = lds[wave][1];
#pragma unroll
  for (int t = 0; t < 8; ++t) {
    const int col = t * 16 + m;
    const float bias = b1[col];
#pragma unroll
    for (int i = 0; i < 8; ++i) {
      const int row = i + base8;
      float v = silu_f(c[t][i] + bias);
      __bf16 h = (__bf16)v;
      h_hi[row * HSTRIDE + col] = h;
      h_lo[row * HSTRIDE + col] = (__bf16)(v - (float)h);
    }
  }
  asm volatile("s_wait_dscnt 0" ::: "memory");  // per-wave LDS RAW fence

  // ---- layer 2: (16 x 128) @ (128 x 128) ----
  v8f d[8];
#pragma unroll
  for (int t = 0; t < 8; ++t) d[t] = zero;
#pragma unroll
  for (int ks = 0; ks < 4; ++ks) {
    const int cb = ks * 32 + base8;
    v8bf h0 = *(const v8bf*)&h_hi[m * HSTRIDE + cb];
    v8bf h1 = *(const v8bf*)&h_hi[m * HSTRIDE + cb + 16];
    v8bf l0 = *(const v8bf*)&h_lo[m * HSTRIDE + cb];
    v8bf l1 = *(const v8bf*)&h_lo[m * HSTRIDE + cb + 16];
    v16bf ah = cat8(h0, h1);
    v16bf al = cat8(l0, l1);
    const __bf16* bh = m2h + (size_t)(ks * 32 + lane) * HDIM;
    const __bf16* bl = m2l + (size_t)(ks * 32 + lane) * HDIM;
#pragma unroll
    for (int t = 0; t < 8; ++t) {
      v16bf wbh = *(const v16bf*)(bh + t * 16);
      v16bf wbl = *(const v16bf*)(bl + t * 16);
      d[t] = mm3(ah, al, wbh, wbl, d[t]);
    }
  }

  // ---- bias + SiLU, write msg, fp32-atomic scatter into agg ----
  int rn[8];
#pragma unroll
  for (int i = 0; i < 8; ++i) {
    const int ee = tile * 16 + i + base8;
    rn[i] = (ee < E) ? recv[ee] : 0;
  }
#pragma unroll
  for (int t = 0; t < 8; ++t) {
    const int col = t * 16 + m;
    const float bias = b2[col];
#pragma unroll
    for (int i = 0; i < 8; ++i) {
      const int ee = tile * 16 + i + base8;
      if (ee < E) {
        float v = silu_f(d[t][i] + bias);
        msgOut[((size_t)b * E + ee) * HDIM + col] = v;
        unsafeAtomicAdd(&agg[((size_t)b * N + rn[i]) * HDIM + col], v);
      }
    }
  }
}

// --------------------------- update kernel ---------------------------------
// One wave per (16-node tile, batch). xn = x + agg/max(cnt,1);
// u = silu(xn @ w_u1 + b_u1); out = xn + u @ w_u2 + b_u2.

__global__ __launch_bounds__(64) void upd_kernel(
    const float* __restrict__ x, const float* __restrict__ agg, const float* __restrict__ cnt,
    const __bf16* __restrict__ u1h, const __bf16* __restrict__ u1l,
    const __bf16* __restrict__ u2h, const __bf16* __restrict__ u2l,
    const float* __restrict__ bu1, const float* __restrict__ bu2,
    float* __restrict__ out, int N, int B_, int numTiles) {
  __shared__ __bf16 lds[2][2][16 * USTRIDE];

  const int wave = threadIdx.x >> 5;
  const int lane = threadIdx.x & 31;
  const int wid  = blockIdx.x * 2 + wave;
  if (wid >= numTiles * B_) return;
  const int tile = wid % numTiles;
  const int b    = wid / numTiles;

  const int m     = lane & 15;
  const int base8 = (lane >> 4) * 8;
  const int node  = tile * 16 + m;
  const int nc    = (node < N) ? node : (N - 1);

  const float* xp = x   + ((size_t)b * N + nc) * HDIM;
  const float* ap = agg + ((size_t)b * N + nc) * HDIM;
  const float  inv = 1.0f / fmaxf(cnt[nc], 1.0f);

  v8f zero = {0.f, 0.f, 0.f, 0.f, 0.f, 0.f, 0.f, 0.f};
  v8f c[16];
#pragma unroll
  for (int t = 0; t < 16; ++t) c[t] = zero;

  // ---- u1: (16 x 128) @ (128 x 256) ----
#pragma unroll
  for (int ks = 0; ks < 4; ++ks) {
    const int cb = ks * 32 + base8;
    float fx[16], fa[16], f[16];
    load16(xp, cb, fx);
    load16(ap, cb, fa);
#pragma unroll
    for (int q = 0; q < 16; ++q) f[q] = fx[q] + fa[q] * inv;
    v16bf ah, al;
    split16(f, ah, al);
    const __bf16* bh = u1h + (size_t)(ks * 32 + lane) * (2 * HDIM);
    const __bf16* bl = u1l + (size_t)(ks * 32 + lane) * (2 * HDIM);
#pragma unroll
    for (int t = 0; t < 16; ++t) {
      v16bf wbh = *(const v16bf*)(bh + t * 16);
      v16bf wbl = *(const v16bf*)(bl + t * 16);
      c[t] = mm3(ah, al, wbh, wbl, c[t]);
    }
  }

  // ---- bias + SiLU, stage u (16 x 256) as bf16 hi/lo ----
  __bf16* u_hi = lds[wave][0];
  __bf16* u_lo = lds[wave][1];
#pragma unroll
  for (int t = 0; t < 16; ++t) {
    const int col = t * 16 + m;
    const float bias = bu1[col];
#pragma unroll
    for (int i = 0; i < 8; ++i) {
      const int row = i + base8;
      float v = silu_f(c[t][i] + bias);
      __bf16 h = (__bf16)v;
      u_hi[row * USTRIDE + col] = h;
      u_lo[row * USTRIDE + col] = (__bf16)(v - (float)h);
    }
  }
  asm volatile("s_wait_dscnt 0" ::: "memory");

  // ---- u2: (16 x 256) @ (256 x 128) ----
  v8f d[8];
#pragma unroll
  for (int t = 0; t < 8; ++t) d[t] = zero;
#pragma unroll
  for (int ks = 0; ks < 8; ++ks) {
    const int cb = ks * 32 + base8;
    v8bf h0 = *(const v8bf*)&u_hi[m * USTRIDE + cb];
    v8bf h1 = *(const v8bf*)&u_hi[m * USTRIDE + cb + 16];
    v8bf l0 = *(const v8bf*)&u_lo[m * USTRIDE + cb];
    v8bf l1 = *(const v8bf*)&u_lo[m * USTRIDE + cb + 16];
    v16bf ah = cat8(h0, h1);
    v16bf al = cat8(l0, l1);
    const __bf16* bh = u2h + (size_t)(ks * 32 + lane) * HDIM;
    const __bf16* bl = u2l + (size_t)(ks * 32 + lane) * HDIM;
#pragma unroll
    for (int t = 0; t < 8; ++t) {
      v16bf wbh = *(const v16bf*)(bh + t * 16);
      v16bf wbl = *(const v16bf*)(bl + t * 16);
      d[t] = mm3(ah, al, wbh, wbl, d[t]);
    }
  }

  // ---- residual: out = xn + u@w_u2 + b_u2 ----
#pragma unroll
  for (int i = 0; i < 8; ++i) {
    const int row = i + base8;
    const int nd  = tile * 16 + row;
    if (nd >= N) continue;
    const float iv = 1.0f / fmaxf(cnt[nd], 1.0f);
    const size_t rowOff = ((size_t)b * N + nd) * HDIM;
#pragma unroll
    for (int t = 0; t < 8; ++t) {
      const int col = t * 16 + m;
      const float xn = x[rowOff + col] + agg[rowOff + col] * iv;
      out[rowOff + col] = xn + d[t][i] + bu2[col];
    }
  }
}

// ------------------------------- launch ------------------------------------

extern "C" void kernel_launch(void* const* d_in, const int* in_sizes, int n_in,
                              void* d_out, int out_size, void* d_ws, size_t ws_size,
                              hipStream_t stream) {
  const float* x    = (const float*)d_in[0];
  const float* ea   = (const float*)d_in[1];
  const int*   send = (const int*)d_in[2];
  const int*   recv = (const int*)d_in[3];
  const float* wm1  = (const float*)d_in[4];
  const float* bm1  = (const float*)d_in[5];
  const float* wm2  = (const float*)d_in[6];
  const float* bm2  = (const float*)d_in[7];
  const float* wu1  = (const float*)d_in[8];
  const float* bu1  = (const float*)d_in[9];
  const float* wu2  = (const float*)d_in[10];
  const float* bu2  = (const float*)d_in[11];

  const int E = in_sizes[2];
  const int H = in_sizes[5];             // == HDIM == 128
  const int B = in_sizes[1] / (E * H);
  const int N = in_sizes[0] / (B * H);

  float* outX   = (float*)d_out;
  float* outMsg = outX + (size_t)B * N * H;

  // workspace carve: agg[B*N*H] f32 | cnt[N] f32 | bf16 hi/lo weight packs
  const long long aggN = (long long)B * N * H;
  float* agg = (float*)d_ws;
  float* cnt = agg + aggN;
  uintptr_t wp = (uintptr_t)(cnt + N);
  wp = (wp + 255) & ~(uintptr_t)255;
  const int n1 = 3 * H * H, n2 = H * H, n3 = 2 * H * H, n4 = 2 * H * H;
  __bf16* m1h = (__bf16*)wp;
  __bf16* m1l = m1h + n1;
  __bf16* m2h = m1l + n1;
  __bf16* m2l = m2h + n2;
  __bf16* u1h = m2l + n2;
  __bf16* u1l = u1h + n3;
  __bf16* u2h = u1l + n3;
  __bf16* u2l = u2h + n4;

  // 1) zero agg + cnt
  {
    long long tot = aggN + N;
    int blocks = (int)((tot + 255) / 256);
    zero_f32<<<blocks, 256, 0, stream>>>(agg, tot);
  }
  // 2) pack weights to bf16 hi/lo
  {
    int tot = n1 + n2 + n3 + n4;
    pack_weights<<<(tot + 255) / 256, 256, 0, stream>>>(
        wm1, wm2, wu1, wu2, m1h, m1l, m2h, m2l, u1h, u1l, u2h, u2l, n1, n2, n3, n4);
  }
  // 3) recv-degree counts
  degree_kernel<<<(E + 255) / 256, 256, 0, stream>>>(recv, cnt, E);
  // 4) message MLP + scatter-add (one wave per 16-edge tile per batch)
  {
    int numTiles = (E + 15) / 16;
    int waves = numTiles * B;
    int blocks = (waves + 3) / 4;
    msg_kernel<<<blocks, 128, 0, stream>>>(x, ea, send, recv, m1h, m1l, m2h, m2l,
                                           bm1, bm2, outMsg, agg, N, E, B, numTiles);
  }
  // 5) node update MLP (one wave per 16-node tile per batch)
  {
    int numTiles = (N + 15) / 16;
    int waves = numTiles * B;
    int blocks = (waves + 1) / 2;
    upd_kernel<<<blocks, 64, 0, stream>>>(x, agg, cnt, u1h, u1l, u2h, u2l,
                                          bu1, bu2, outX, N, B, numTiles);
  }
}
